// TriangleMultiplication_25829933318696
// MI455X (gfx1250) — matrix operand order repaired and checked by reading, not compiled, not run on hardware
//
#include <hip/hip_runtime.h>
#include <hip/hip_bf16.h>

// ---------------------------------------------------------------------------
// Triangle multiplicative update (outgoing), N=512, c_z=c_h=128.
// Dominant cost: einsum 'ikc,jkc->ijc' = 128 x [512x512x512] GEMMs = 34.4 GFLOP
// -> bf16 WMMA (v_wmma_f32_16x16x32_bf16). a/b stored bf16 [c][i][k] (64MB each,
// L2-resident on the 192MB L2), x fp32 [i][j][c] (128MB). ws usage = 256MB.
// Einsum kernel: 4 waves/block, 64x64 tile per wave (16 accums, 128 VGPRs)
// -> 16 WMMAs per 8KB of fragment loads (512 B/WMMA register-feed intensity).
// ---------------------------------------------------------------------------

#define NDIM 512
#define CDIM 128
#define NN   (512 * 512)
#define LDZ  136   // padded LDS row stride in bf16 elems (272B, multiple of 16B)

typedef __attribute__((ext_vector_type(16))) __bf16 v16bf;
typedef __attribute__((ext_vector_type(8)))  float  v8f;

union Frag { v16bf v; uint4 u[2]; };

// A-fragment (16x32 bf16): lane m = lane&15 is the row, half = lane>>4 selects
// K-chunks [8h..8h+7] and [16+8h..16+8h+7] (ISA 7.12.2, 16-bit A 16x32).
__device__ inline v16bf load_a_frag(const __bf16* base, int stride, int k0, int lane) {
  int m = lane & 15, h = lane >> 4;
  const __bf16* p = base + (size_t)m * stride + k0;
  Frag f;
  f.u[0] = *(const uint4*)(p + 8 * h);
  f.u[1] = *(const uint4*)(p + 16 + 8 * h);
  return f.v;
}

// B-fragment (32x16 bf16): lane n = lane&15 is the column, half selects
// K-chunks [8h..8h+7] and [16+8h..16+8h+7]. base points at column n0's row (row-major
// [n][k] storage, i.e. B^T with K contiguous).
__device__ inline v16bf load_b_frag(const __bf16* base, int stride, int k0, int lane) {
  int n = lane & 15, h = lane >> 4;
  const __bf16* p = base + (size_t)n * stride + k0;
  Frag f;
  f.u[0] = *(const uint4*)(p + 8 * h);
  f.u[1] = *(const uint4*)(p + 16 + 8 * h);
  return f.v;
}

// LayerNorm 128 rows x 128 ch -> bf16 LDS tile [128][LDZ]. 2 threads per row.
__device__ inline void layernorm_to_lds(const float* __restrict__ src,
                                        const float* __restrict__ sc,
                                        const float* __restrict__ bi,
                                        __bf16* dst, int tid) {
  int r = tid >> 1, h = tid & 1;
  const float* row = src + (size_t)r * CDIM + h * 64;
  float s = 0.f, ss = 0.f;
#pragma unroll
  for (int c = 0; c < 64; c += 4) {
    float4 v = *(const float4*)(row + c);
    s  += v.x + v.y + v.z + v.w;
    ss += v.x * v.x + v.y * v.y + v.z * v.z + v.w * v.w;
  }
  s  += __shfl_xor(s, 1, 32);
  ss += __shfl_xor(ss, 1, 32);
  float mu   = s * (1.f / 128.f);
  float var  = ss * (1.f / 128.f) - mu * mu;
  float rstd = rsqrtf(var + 1e-5f);
#pragma unroll
  for (int c = 0; c < 64; c += 4) {
    float4 v = *(const float4*)(row + c);
    int cc = h * 64 + c;
    dst[r * LDZ + cc + 0] = (__bf16)((v.x - mu) * rstd * sc[cc + 0] + bi[cc + 0]);
    dst[r * LDZ + cc + 1] = (__bf16)((v.y - mu) * rstd * sc[cc + 1] + bi[cc + 1]);
    dst[r * LDZ + cc + 2] = (__bf16)((v.z - mu) * rstd * sc[cc + 2] + bi[cc + 2]);
    dst[r * LDZ + cc + 3] = (__bf16)((v.w - mu) * rstd * sc[cc + 3] + bi[cc + 3]);
  }
}

// Stage W[k][n] (128x128 fp32) as transposed bf16 Wt[n][k] in LDS.
__device__ inline void load_w_t(const float* __restrict__ w, __bf16* wt, int tid) {
  for (int idx = tid; idx < CDIM * CDIM; idx += 256) {
    int k = idx >> 7, n = idx & 127;
    wt[n * LDZ + k] = (__bf16)w[idx];
  }
}

// One wave: 16 rows of zn (K=128) x Wt(128x128) -> 8 accumulator tiles (16x16).
__device__ inline void gemm_row16_x128(const __bf16* zn_rows, const __bf16* wt,
                                       int lane, v8f acc[8]) {
#pragma unroll
  for (int k0 = 0; k0 < CDIM; k0 += 32) {
    v16bf af = load_a_frag(zn_rows, LDZ, k0, lane);
#pragma unroll
    for (int nt = 0; nt < 8; ++nt) {
      v16bf bf = load_b_frag(wt + nt * 16 * LDZ, LDZ, k0, lane);
      acc[nt] = __builtin_amdgcn_wmma_f32_16x16x32_bf16(
          false, af, false, bf, (short)0, acc[nt], false, false);
    }
  }
}

__device__ inline float sigm(float x) { return 1.f / (1.f + __expf(-x)); }

// ---------------------------------------------------------------------------
// Kernel A: LN(z) + gated left/right projections -> a,b in bf16 [c][i][k].
// ---------------------------------------------------------------------------
__global__ __launch_bounds__(256) void fused_gate_proj(
    const float* __restrict__ z,       const float* __restrict__ mask,
    const float* __restrict__ ln_s,    const float* __restrict__ ln_b,
    const float* __restrict__ w_left,  const float* __restrict__ b_left,
    const float* __restrict__ w_right, const float* __restrict__ b_right,
    const float* __restrict__ w_lgate, const float* __restrict__ b_lgate,
    const float* __restrict__ w_rgate, const float* __restrict__ b_rgate,
    __bf16* __restrict__ Aout, __bf16* __restrict__ Bout)
{
  __shared__ __bf16 zn[128 * LDZ];
  __shared__ __bf16 wt[128 * LDZ];
  __shared__ float  rmask[128];

  int tid = threadIdx.x, lane = tid & 31, wave = tid >> 5;
  size_t m0 = (size_t)blockIdx.x * 128;

  layernorm_to_lds(z + m0 * CDIM, ln_s, ln_b, zn, tid);
  if ((tid & 1) == 0) rmask[tid >> 1] = mask[m0 + (tid >> 1)];
  __syncthreads();

  const __bf16* myrows = zn + wave * 16 * LDZ;
  int nl = lane & 15, h = lane >> 4;

#pragma unroll 1
  for (int pair = 0; pair < 2; ++pair) {
    const float* wp = pair ? w_right : w_left;
    const float* bp = pair ? b_right : b_left;
    const float* wg = pair ? w_rgate : w_lgate;
    const float* bg = pair ? b_rgate : b_lgate;
    __bf16* outp = pair ? Bout : Aout;

    __syncthreads();                 // prior readers of wt are done
    load_w_t(wp, wt, tid);
    __syncthreads();
    v8f accP[8] = {};
    gemm_row16_x128(myrows, wt, lane, accP);
    __syncthreads();
    load_w_t(wg, wt, tid);
    __syncthreads();
    v8f accG[8] = {};
    gemm_row16_x128(myrows, wt, lane, accG);

    // a = mask * sigmoid(zn@Wg + bg) * (zn@Wp + bp), stored [c][i*N+k] bf16
#pragma unroll
    for (int nt = 0; nt < 8; ++nt) {
      int n = nt * 16 + nl;
      float bpv = bp[n], bgv = bg[n];
      __bf16* obase = outp + (size_t)n * NN + m0 + wave * 16 + 8 * h;
#pragma unroll
      for (int v = 0; v < 8; ++v) {
        float p  = accP[nt][v] + bpv;
        float g  = accG[nt][v] + bgv;
        float mk = rmask[wave * 16 + 8 * h + v];
        obase[v] = (__bf16)(mk * p * sigm(g));
      }
    }
  }
}

// ---------------------------------------------------------------------------
// Kernel B: x[i,j,c] = sum_k a[i,k,c]*b[j,k,c]  ==  X_c = A_c * B_c^T.
// Block = 128 threads (4 waves, 2x2), each wave owns a 64x64 tile:
// 16 f32 accumulators (128 VGPRs) + 4 A-frags + 4 B-frags per K-step.
// Operands are L2-resident (64MB each vs 192MB L2); fragments load straight
// from global as contiguous b128 pairs.
// ---------------------------------------------------------------------------
__global__ __launch_bounds__(128) void triangle_einsum(
    const __bf16* __restrict__ A, const __bf16* __restrict__ B,
    float* __restrict__ X)
{
  int c  = blockIdx.z;
  int i0 = blockIdx.y * 128, j0 = blockIdx.x * 128;
  const __bf16* Ab = A + (size_t)c * NN;
  const __bf16* Bb = B + (size_t)c * NN;

  int tid = threadIdx.x, lane = tid & 31, wave = tid >> 5;
  int wm = (wave >> 1) * 64;   // 0 or 64
  int wn = (wave & 1) * 64;    // 0 or 64

  v8f acc[4][4] = {};
#pragma unroll 1
  for (int k0 = 0; k0 < NDIM; k0 += 32) {
    if (k0 + 32 < NDIM)        // pull next K-chunk toward L0/L2
      __builtin_prefetch(Ab + (size_t)(i0 + wm + (lane & 15)) * NDIM + k0 + 32, 0, 1);

    v16bf af[4], bf[4];
#pragma unroll
    for (int mt = 0; mt < 4; ++mt)
      af[mt] = load_a_frag(Ab + (size_t)(i0 + wm + mt * 16) * NDIM, NDIM, k0, lane);
#pragma unroll
    for (int nt = 0; nt < 4; ++nt)
      bf[nt] = load_b_frag(Bb + (size_t)(j0 + wn + nt * 16) * NDIM, NDIM, k0, lane);
#pragma unroll
    for (int mt = 0; mt < 4; ++mt)
#pragma unroll
      for (int nt = 0; nt < 4; ++nt)
        acc[mt][nt] = __builtin_amdgcn_wmma_f32_16x16x32_bf16(
            false, af[mt], false, bf[nt], (short)0, acc[mt][nt], false, false);
  }

  int nl = lane & 15, h = lane >> 4;
#pragma unroll
  for (int mt = 0; mt < 4; ++mt)
#pragma unroll
    for (int nt = 0; nt < 4; ++nt) {
      int gj = j0 + wn + nt * 16 + nl;
#pragma unroll
      for (int v = 0; v < 8; ++v) {
        int gi = i0 + wm + mt * 16 + v + 8 * h;
        X[((size_t)gi * NDIM + gj) * CDIM + c] = acc[mt][nt][v];
      }
    }
}

// ---------------------------------------------------------------------------
// Kernel C: out = (LN(x) @ w_out + b_out) * sigmoid(LN(z) @ w_gate + b_gate)
// ---------------------------------------------------------------------------
__global__ __launch_bounds__(256) void fused_output(
    const float* __restrict__ x,        const float* __restrict__ z,
    const float* __restrict__ ln_in_s,  const float* __restrict__ ln_in_b,
    const float* __restrict__ ln_out_s, const float* __restrict__ ln_out_b,
    const float* __restrict__ w_out,    const float* __restrict__ b_out,
    const float* __restrict__ w_gate,   const float* __restrict__ b_gate,
    float* __restrict__ out)
{
  __shared__ __bf16 tn[128 * LDZ];
  __shared__ __bf16 wt[128 * LDZ];

  int tid = threadIdx.x, lane = tid & 31, wave = tid >> 5;
  size_t m0 = (size_t)blockIdx.x * 128;
  const __bf16* myrows = tn + wave * 16 * LDZ;
  int nl = lane & 15, h = lane >> 4;

  // gate path: LN(z) @ w_gate
  layernorm_to_lds(z + m0 * CDIM, ln_in_s, ln_in_b, tn, tid);
  load_w_t(w_gate, wt, tid);
  __syncthreads();
  v8f accG[8] = {};
  gemm_row16_x128(myrows, wt, lane, accG);
  __syncthreads();

  // output path: LN(x) @ w_out   (reuse tn/wt)
  layernorm_to_lds(x + m0 * CDIM, ln_out_s, ln_out_b, tn, tid);
  load_w_t(w_out, wt, tid);
  __syncthreads();
  v8f accO[8] = {};
  gemm_row16_x128(myrows, wt, lane, accO);

#pragma unroll
  for (int nt = 0; nt < 8; ++nt) {
    int n = nt * 16 + nl;
    float bo = b_out[n], bg = b_gate[n];
    float* obase = out + (m0 + wave * 16 + 8 * h) * CDIM + n;
#pragma unroll
    for (int v = 0; v < 8; ++v) {
      float o = accO[nt][v] + bo;
      float g = accG[nt][v] + bg;
      obase[(size_t)v * CDIM] = o * sigm(g);
    }
  }
}

// ---------------------------------------------------------------------------
extern "C" void kernel_launch(void* const* d_in, const int* in_sizes, int n_in,
                              void* d_out, int out_size, void* d_ws, size_t ws_size,
                              hipStream_t stream) {
  const float* z        = (const float*)d_in[0];
  const float* mask     = (const float*)d_in[1];
  const float* ln_in_s  = (const float*)d_in[2];
  const float* ln_in_b  = (const float*)d_in[3];
  const float* w_left   = (const float*)d_in[4];
  const float* b_left   = (const float*)d_in[5];
  const float* w_right  = (const float*)d_in[6];
  const float* b_right  = (const float*)d_in[7];
  const float* w_lgate  = (const float*)d_in[8];
  const float* b_lgate  = (const float*)d_in[9];
  const float* w_rgate  = (const float*)d_in[10];
  const float* b_rgate  = (const float*)d_in[11];
  const float* ln_out_s = (const float*)d_in[12];
  const float* ln_out_b = (const float*)d_in[13];
  const float* w_out    = (const float*)d_in[14];
  const float* b_out    = (const float*)d_in[15];
  const float* w_gate   = (const float*)d_in[16];
  const float* b_gate   = (const float*)d_in[17];

  char* ws = (char*)d_ws;
  __bf16* a_ws = (__bf16*)(ws);                                  //  64 MB
  __bf16* b_ws = (__bf16*)(ws + (size_t)CDIM * NN * 2);          //  64 MB
  float*  x_ws = (float*) (ws + (size_t)2 * CDIM * NN * 2);      // 128 MB

  fused_gate_proj<<<dim3(NN / 128), 256, 0, stream>>>(
      z, mask, ln_in_s, ln_in_b, w_left, b_left, w_right, b_right,
      w_lgate, b_lgate, w_rgate, b_rgate, a_ws, b_ws);

  triangle_einsum<<<dim3(NDIM / 128, NDIM / 128, CDIM), 128, 0, stream>>>(
      a_ws, b_ws, x_ws);

  fused_output<<<dim3(NN / 128), 256, 0, stream>>>(
      x_ws, z, ln_in_s, ln_in_b, ln_out_s, ln_out_b,
      w_out, b_out, w_gate, b_gate, (float*)d_out);
}
